// TopicSNE_17583596110565
// MI455X (gfx1250) — matrix operand order repaired and checked by reading, not compiled, not run on hardware
//
#include <hip/hip_runtime.h>
#include <hip/hip_bf16.h>

typedef _Float16 v16h __attribute__((ext_vector_type(16)));
typedef _Float16 v8h  __attribute__((ext_vector_type(8)));
typedef float    v8f  __attribute__((ext_vector_type(8)));

#define N_POINTS 8192
#define N_TOPICS 64
#define B_PAIRS  524288

// Fast reciprocal: v_rcp_f32 + one Newton-Raphson step (~fp32 accuracy,
// avoids the ~10-instruction IEEE division sequence per element).
__device__ __forceinline__ float fast_rcp(float x) {
    float y = __builtin_amdgcn_rcpf(x);
    float e = fmaf(-x, y, 1.0f);
    return fmaf(y, e, y);
}

// ---------------------------------------------------------------------------
// Kernel 1: zero accumulators, convert W -> f16, compute row norms sq[r].
// Grid covers exactly N_POINTS*N_TOPICS = 524288 elements (2048 x 256).
// ---------------------------------------------------------------------------
__global__ void __launch_bounds__(256)
prep_kernel(const float* __restrict__ W, _Float16* __restrict__ Wh,
            float* __restrict__ sq, double* __restrict__ acc) {
    int tid = blockIdx.x * 256 + threadIdx.x;
    if (tid == 0) { acc[0] = 0.0; acc[1] = 0.0; acc[2] = 0.0; }
    Wh[tid] = (_Float16)W[tid];
    if (tid < N_POINTS) {
        const float* row = W + tid * N_TOPICS;
        float s = 0.0f;
#pragma unroll
        for (int k = 0; k < N_TOPICS; ++k) { float v = row[k]; s = fmaf(v, v, s); }
        sq[tid] = s;
    }
}

// ---------------------------------------------------------------------------
// Kernel 2: Gram-matrix partition sum via WMMA f16 (f32 accumulate).
// One wave32 per 16x16 tile of G = W * W^T; 8 waves per block.
// Symmetry: only tiles with tileRow <= tileCol computed; off-diagonal
// tiles weighted x2 (the skip branch is wave-uniform -> EXEC stays all-1s).
// d2 = sq[i] + sq[j] - 2*G[i][j];  acc[0] += sum 1/(1+d2).
// ---------------------------------------------------------------------------
__global__ void __launch_bounds__(256)
gram_part_kernel(const _Float16* __restrict__ Wh, const float* __restrict__ sq,
                 double* __restrict__ acc) {
    const int lane = threadIdx.x & 31;
    const int wave = threadIdx.x >> 5;
    const int t    = blockIdx.x * 8 + wave;          // tile id, 0 .. 512*512-1
    const int tr   = t >> 9;                         // tile row 0..511
    const int tc   = t & 511;                        // tile col 0..511
    if (tr > tc) return;                             // wave-uniform skip (symmetry)

    const int rowBase = tr << 4;
    const int colBase = tc << 4;
    const int m  = lane & 15;
    const int hi = lane >> 4;

    // A fragment: row (rowBase+m) of W; ISA 16-bit A layout (16x32, MxK):
    //   elements 0..7  <- K = kc + hi*8      .. +7
    //   elements 8..15 <- K = kc + 16 + hi*8 .. +7
    // B fragment: for a Gram matrix, B[k][n] = W[colBase+n][k]; ISA B layout:
    //   lane n = lane&15, same K order as A (0..7 <- kc+hi*8.., 8..15 <- kc+16+hi*8..)
    const _Float16* aRow = Wh + (rowBase + m) * N_TOPICS;
    const _Float16* bRow = Wh + (colBase + m) * N_TOPICS;

    v8f c = {};
#pragma unroll
    for (int kc = 0; kc < N_TOPICS; kc += 32) {
        v8h alo = *(const v8h*)(aRow + kc + hi * 8);
        v8h ahi = *(const v8h*)(aRow + kc + 16 + hi * 8);
        v16h a;
#pragma unroll
        for (int e = 0; e < 8; ++e) { a[e] = alo[e]; a[8 + e] = ahi[e]; }
        v8h blo = *(const v8h*)(bRow + kc + hi * 8);
        v8h bhi = *(const v8h*)(bRow + kc + 16 + hi * 8);
        v16h b;
#pragma unroll
        for (int e = 0; e < 8; ++e) { b[e] = blo[e]; b[8 + e] = bhi[e]; }
        c = __builtin_amdgcn_wmma_f32_16x16x32_f16(
                /*neg_a=*/false, a, /*neg_b=*/false, b,
                /*c_mod=*/(short)0, c, /*reuse_a=*/false, /*reuse_b=*/false);
    }

    // C/D layout: VGPR r: lanes 0-15 -> M=r, N=lane; lanes 16-31 -> M=r+8, N=lane-16
    const float sqn  = sq[colBase + m];
    const int mBase  = rowBase + hi * 8;
    float partial = 0.0f;
#pragma unroll
    for (int r = 0; r < 8; ++r) {
        float d2 = sq[mBase + r] + sqn - 2.0f * c[r];
        d2 = fmaxf(d2, 0.0f);
        partial += fast_rcp(1.0f + d2);
    }
    if (tr != tc) partial *= 2.0f;                   // symmetric counterpart
#pragma unroll
    for (int off = 16; off; off >>= 1) partial += __shfl_xor(partial, off, 32);
    if (lane == 0) atomicAdd(&acc[0], (double)partial);
}

// ---------------------------------------------------------------------------
// Kernel 3: pair term.  S1 = sum pij*(log pij + log s),  S2 = sum pij,
// where s = N_TOPICS + ||xi - xj||^2.  Gathers hit the L2-resident W table.
// ---------------------------------------------------------------------------
__global__ void __launch_bounds__(256)
pair_kernel(const float* __restrict__ pij, const int* __restrict__ I,
            const int* __restrict__ J, const float* __restrict__ W,
            double* __restrict__ acc) {
    int tid = blockIdx.x * 256 + threadIdx.x;
    float p = pij[tid];
    const float4* xi = (const float4*)(W + (long)I[tid] * N_TOPICS);
    const float4* xj = (const float4*)(W + (long)J[tid] * N_TOPICS);
    float s = (float)N_TOPICS;
#pragma unroll
    for (int k = 0; k < N_TOPICS / 4; ++k) {
        float4 a = xi[k], b = xj[k];
        float dx = a.x - b.x, dy = a.y - b.y, dz = a.z - b.z, dw = a.w - b.w;
        s = fmaf(dx, dx, s); s = fmaf(dy, dy, s);
        s = fmaf(dz, dz, s); s = fmaf(dw, dw, s);
    }
    float t1 = p * (logf(p) + logf(s));
    float t2 = p;
#pragma unroll
    for (int off = 16; off; off >>= 1) {
        t1 += __shfl_xor(t1, off, 32);
        t2 += __shfl_xor(t2, off, 32);
    }
    if ((threadIdx.x & 31) == 0) {
        atomicAdd(&acc[1], (double)t1);
        atomicAdd(&acc[2], (double)t2);
    }
}

// ---------------------------------------------------------------------------
// Kernel 4: finalize.  part = acc[0] - n;  loss = S1 + log(part)*S2.
// ---------------------------------------------------------------------------
__global__ void finalize_kernel(const double* __restrict__ acc, float* __restrict__ out) {
    double part = acc[0] - (double)N_POINTS;
    out[0] = (float)(acc[1] + log(part) * acc[2]);
}

extern "C" void kernel_launch(void* const* d_in, const int* in_sizes, int n_in,
                              void* d_out, int out_size, void* d_ws, size_t ws_size,
                              hipStream_t stream) {
    const float* pij = (const float*)d_in[0];
    const int*   I   = (const int*)d_in[1];
    const int*   J   = (const int*)d_in[2];
    const float* W   = (const float*)d_in[3];

    char* ws = (char*)d_ws;
    double*   acc = (double*)ws;                                     // 3 doubles
    float*    sq  = (float*)(ws + 32);                               // 8192 f32
    _Float16* Wh  = (_Float16*)(ws + 32 + N_POINTS * sizeof(float)); // 1 MB f16

    prep_kernel<<<(N_POINTS * N_TOPICS) / 256, 256, 0, stream>>>(W, Wh, sq, acc);
    gram_part_kernel<<<(512 * 512) / 8, 256, 0, stream>>>(Wh, sq, acc);
    pair_kernel<<<B_PAIRS / 256, 256, 0, stream>>>(pij, I, J, W, acc);
    finalize_kernel<<<1, 1, 0, stream>>>(acc, (float*)d_out);
}
